// MLPAttention_55336358642140
// MI455X (gfx1250) — matrix operand order repaired and checked by reading, not compiled, not run on hardware
//
#include <hip/hip_runtime.h>
#include <hip/hip_bf16.h>

#define SRC_LEN 2048
#define BSZ     64
#define CTX_DIM 512
#define DEC_DIM 512
#define ATT_DIM 256

typedef __attribute__((ext_vector_type(16))) __bf16 v16bf;
typedef __attribute__((ext_vector_type(8)))  float  v8f;

__device__ __forceinline__ unsigned short f2bf(float f) {
    unsigned int u = __float_as_uint(f);
    u += 0x7FFFu + ((u >> 16) & 1u);        // round-to-nearest-even
    return (unsigned short)(u >> 16);
}

// ---------------------------------------------------------------- prep ----
__global__ void mlpattn_wenc_bf16(const float* __restrict__ W_enc,
                                  unsigned short* __restrict__ Wb) {
    int i = blockIdx.x * blockDim.x + threadIdx.x;
    if (i < ATT_DIM * CTX_DIM) Wb[i] = f2bf(W_enc[i]);
}

__global__ void mlpattn_decproj(const float* __restrict__ dec_state,
                                const float* __restrict__ W_dec,
                                const float* __restrict__ b_enc,
                                float* __restrict__ decproj) {
    int i = blockIdx.x * blockDim.x + threadIdx.x;      // b*256 + a
    if (i >= BSZ * ATT_DIM) return;
    int b = i >> 8, a = i & 255;
    const float* ds = dec_state + b * DEC_DIM;
    const float* wd = W_dec + (size_t)a * DEC_DIM;
    float acc = 0.f;
    for (int d = 0; d < DEC_DIM; d += 4)
        acc += ds[d] * wd[d] + ds[d+1] * wd[d+1] + ds[d+2] * wd[d+2] + ds[d+3] * wd[d+3];
    decproj[i] = acc + b_enc[a];
}

__global__ void mlpattn_zero(float* __restrict__ p, int n) {
    int i = blockIdx.x * blockDim.x + threadIdx.x;
    if (i < n) p[i] = 0.f;
}

// ------------------------------------------------------- WMMA scores ------
// One wave per 16-row M tile of the (131072 x 512) x (512 x 256) GEMM,
// fused with tanh(.)·v reduction producing attn_scores[s*64+b].
__global__ void __launch_bounds__(256)
mlpattn_scores_wmma(const float* __restrict__ src,            // [S*B, 512] f32
                    const unsigned short* __restrict__ Wb,    // [256, 512] bf16
                    const float* __restrict__ decproj,        // [64, 256]
                    const float* __restrict__ vvec,           // [256]
                    float* __restrict__ scores) {             // [S*B]
    __shared__ __attribute__((aligned(64))) unsigned short Wlds[ATT_DIM * 32]; // 16KB K-slice [n][kk]

    const int wave = threadIdx.x >> 5;
    const int lane = threadIdx.x & 31;
    const int tile = blockIdx.x * 8 + wave;
    const int r0   = tile * 16;             // rows r = s*64 + b ; tile never crosses s
    const int b0   = r0 & (BSZ - 1);
    const int m    = lane & 15;             // A row (M), and C/D column (N) index
    const int half = lane >> 4;

    v8f acc[16];
    v8f zero = {};
#pragma unroll
    for (int na = 0; na < 16; ++na) acc[na] = zero;

    const float* arow = src + (size_t)(r0 + m) * CTX_DIM;

    for (int kb = 0; kb < CTX_DIM / 32; ++kb) {
        __syncthreads();
        {   // stage bf16 W slice: Wlds[n*32 + kk] = Wb[n*512 + kb*32 + kk]
            const uint4* g = (const uint4*)(Wb + (size_t)threadIdx.x * CTX_DIM + kb * 32);
            uint4* l = (uint4*)&Wlds[threadIdx.x * 32];
            l[0] = g[0]; l[1] = g[1]; l[2] = g[2]; l[3] = g[3];
        }
        __syncthreads();

        // Prefetch the next K-slice of this wave's A rows (global_prefetch_b8)
        if (kb + 1 < CTX_DIM / 32)
            __builtin_prefetch(arow + (kb + 1) * 32 + half * 8, 0, 0);

        // A fragment, contiguous k order (this code's choice, not AMD's table order): lane half 0 -> K 0-15 ; half 1 -> K 16-31
        // Native f32->bf16 casts so the backend can use packed v_cvt_pk_bf16_f32.
        const float4* p0 = (const float4*)(arow + kb * 32 + half * 16);
        const float4* p1 = (const float4*)(arow + kb * 32 + half * 16 + 8);
        float4 q0 = p0[0], q1 = p0[1], q2 = p1[0], q3 = p1[1];
        v16bf af;
        af[0]  = (__bf16)q0.x; af[1]  = (__bf16)q0.y; af[2]  = (__bf16)q0.z; af[3]  = (__bf16)q0.w;
        af[4]  = (__bf16)q1.x; af[5]  = (__bf16)q1.y; af[6]  = (__bf16)q1.z; af[7]  = (__bf16)q1.w;
        af[8]  = (__bf16)q2.x; af[9]  = (__bf16)q2.y; af[10] = (__bf16)q2.z; af[11] = (__bf16)q2.w;
        af[12] = (__bf16)q3.x; af[13] = (__bf16)q3.y; af[14] = (__bf16)q3.z; af[15] = (__bf16)q3.w;

#pragma unroll
        for (int na = 0; na < 16; ++na) {
            // B, same contiguous k order as A: lane = column n, lanes 0-15 hold K 0-15, 16-31 hold K 16-31
            const v16bf* bp = (const v16bf*)&Wlds[(na * 16 + m) * 32 + half * 16];
            acc[na] = __builtin_amdgcn_wmma_f32_16x16x32_bf16(
                false, af, false, *bp, (short)0, acc[na], false, false);
        }
    }

    // Fused epilogue: score partials per row; C/D layout: VGPR j = row j + 8*half, col = m
    float part[8];
#pragma unroll
    for (int j = 0; j < 8; ++j) part[j] = 0.f;
#pragma unroll
    for (int na = 0; na < 16; ++na) {
        int n = na * 16 + m;
        float vn = vvec[n];
#pragma unroll
        for (int j = 0; j < 8; ++j) {
            int row_b = b0 + j + 8 * half;
            float h = tanhf(acc[na][j] + decproj[row_b * ATT_DIM + n]);
            part[j] += h * vn;
        }
    }
#pragma unroll
    for (int j = 0; j < 8; ++j) {
        part[j] += __shfl_xor(part[j], 1, 32);
        part[j] += __shfl_xor(part[j], 2, 32);
        part[j] += __shfl_xor(part[j], 4, 32);
        part[j] += __shfl_xor(part[j], 8, 32);
    }
    if (m == 0) {
#pragma unroll
        for (int j = 0; j < 8; ++j) scores[r0 + j + 8 * half] = part[j];
    }
}

// ----------------------------------------------------------- softmax ------
__global__ void mlpattn_softmax(const float* __restrict__ scores,
                                const unsigned char* __restrict__ mask,
                                float* __restrict__ norm) {
    int b = blockIdx.x, tid = threadIdx.x;
    __shared__ float red[256];
    float mx = -3.4e38f;
    for (int s = tid; s < SRC_LEN; s += 256)
        if (!mask[s * BSZ + b]) mx = fmaxf(mx, scores[s * BSZ + b]);
    red[tid] = mx; __syncthreads();
    for (int off = 128; off > 0; off >>= 1) {
        if (tid < off) red[tid] = fmaxf(red[tid], red[tid + off]);
        __syncthreads();
    }
    mx = red[0]; __syncthreads();
    float sum = 0.f;
    for (int s = tid; s < SRC_LEN; s += 256) {
        int idx = s * BSZ + b;
        float e = mask[idx] ? 0.f : expf(scores[idx] - mx);
        norm[idx] = e;
        sum += e;
    }
    red[tid] = sum; __syncthreads();
    for (int off = 128; off > 0; off >>= 1) {
        if (tid < off) red[tid] += red[tid + off];
        __syncthreads();
    }
    float inv = 1.f / red[0];
    for (int s = tid; s < SRC_LEN; s += 256) norm[s * BSZ + b] *= inv;
}

// ------------------------------------------------- weighted context -------
__global__ void mlpattn_context(const float* __restrict__ src,
                                const float* __restrict__ norm,
                                float* __restrict__ out) {
    int b = blockIdx.x;
    int s0 = blockIdx.y * 64;
    int c = threadIdx.x;                    // handles c and c+256
    float a0 = 0.f, a1 = 0.f;
    for (int s = s0; s < s0 + 64; ++s) {
        float w = norm[s * BSZ + b];
        const float* row = src + ((size_t)s * BSZ + b) * CTX_DIM;
        a0 += row[c] * w;
        a1 += row[c + 256] * w;
    }
    atomicAdd(&out[b * CTX_DIM + c], a0);
    atomicAdd(&out[b * CTX_DIM + c + 256], a1);
}

// ------------------------------------------------------------ launch ------
extern "C" void kernel_launch(void* const* d_in, const int* in_sizes, int n_in,
                              void* d_out, int out_size, void* d_ws, size_t ws_size,
                              hipStream_t stream) {
    const float*         decoder_state = (const float*)d_in[0];
    const float*         source_hids   = (const float*)d_in[1];
    const unsigned char* mask          = (const unsigned char*)d_in[2];
    const float*         W_enc         = (const float*)d_in[3];
    const float*         b_enc         = (const float*)d_in[4];
    const float*         W_dec         = (const float*)d_in[5];
    const float*         v             = (const float*)d_in[6];

    float* out_ctx  = (float*)d_out;                    // [64, 512]
    float* out_norm = out_ctx + BSZ * CTX_DIM;          // [2048, 64]

    unsigned short* Wb = (unsigned short*)d_ws;                         // 256 KB
    float* decproj = (float*)((char*)d_ws + ATT_DIM * CTX_DIM * 2);     // 64 KB
    float* scores  = decproj + BSZ * ATT_DIM;                           // 512 KB

    mlpattn_wenc_bf16<<<(ATT_DIM * CTX_DIM + 255) / 256, 256, 0, stream>>>(W_enc, Wb);
    mlpattn_decproj<<<(BSZ * ATT_DIM + 255) / 256, 256, 0, stream>>>(decoder_state, W_dec, b_enc, decproj);
    mlpattn_zero<<<(BSZ * CTX_DIM + 255) / 256, 256, 0, stream>>>(out_ctx, BSZ * CTX_DIM);
    mlpattn_scores_wmma<<<(SRC_LEN * BSZ / 16) / 8, 256, 0, stream>>>(source_hids, Wb, decproj, v, scores);
    mlpattn_softmax<<<BSZ, 256, 0, stream>>>(scores, mask, out_norm);
    mlpattn_context<<<dim3(BSZ, SRC_LEN / 64), 256, 0, stream>>>(source_hids, out_norm, out_ctx);
}
